// NestedBlock_16183436771520
// MI455X (gfx1250) — matrix operand order repaired and checked by reading, not compiled, not run on hardware
//
#include <hip/hip_runtime.h>
#include <stddef.h>

// ---------------------------------------------------------------------------
// Nested-MoE transformer block for MI455X (gfx1250, wave32, WMMA + TDM).
// All dense math runs through v_wmma_f32_16x16x32_f16 (f16 operands, f32 acc).
// GEMM operand tiles are staged into LDS by the Tensor Data Mover with a
// double-buffered pipeline (tensor_load_to_lds + s_wait_tensorcnt), so the
// DMA for tile i+1 overlaps the WMMAs on tile i across all 8 waves.
// Problem sizes (fixed): B=4, N=1024, D=1024, E=3, H=16, hd=64, DFF=4096.
// ---------------------------------------------------------------------------

typedef __attribute__((ext_vector_type(16))) _Float16 v16h;
typedef __attribute__((ext_vector_type(8)))  _Float16 v8h;
typedef __attribute__((ext_vector_type(8)))  float    v8f;
typedef __attribute__((ext_vector_type(4)))  unsigned int u32x4;
typedef __attribute__((ext_vector_type(8)))  unsigned int u32x8;

#define SHUF16(lo, hi) \
  __builtin_shufflevector((lo), (hi), 0,1,2,3,4,5,6,7,8,9,10,11,12,13,14,15)

static constexpr size_t TOK = 4096;   // B*N
static constexpr size_t Dm  = 1024;
static constexpr size_t DFF = 4096;

// ---- workspace layout (bytes); total ~134 MB -------------------------------
static constexpr size_t OFF_MTOK = 0;
static constexpr size_t OFF_PSEL = OFF_MTOK + TOK * 4;
static constexpr size_t OFF_H16  = 32768;
static constexpr size_t OFF_WQT  = OFF_H16 + TOK * Dm * 2;
static constexpr size_t OFF_WKT  = OFF_WQT + Dm * Dm * 2;
static constexpr size_t OFF_WVT  = OFF_WKT + Dm * Dm * 2;
static constexpr size_t OFF_WOT  = OFF_WVT + Dm * Dm * 2;
static constexpr size_t OFF_W1T  = OFF_WOT + Dm * Dm * 2;
static constexpr size_t OFF_W2T  = OFF_W1T + Dm * DFF * 2;
static constexpr size_t OFF_Q16  = OFF_W2T + DFF * Dm * 2;
static constexpr size_t OFF_K16  = OFF_Q16 + TOK * Dm * 2;
static constexpr size_t OFF_V16  = OFF_K16 + TOK * Dm * 2;
static constexpr size_t OFF_VT16 = OFF_V16 + TOK * Dm * 2;
static constexpr size_t OFF_O16  = OFF_VT16 + TOK * Dm * 2;
static constexpr size_t OFF_Z    = OFF_O16 + TOK * Dm * 2;
static constexpr size_t OFF_H2   = OFF_Z + TOK * Dm * 4;
static constexpr size_t OFF_U16  = OFF_H2 + TOK * Dm * 2;

// ---------------------------------------------------------------------------
// Tensor DMA Descriptor (D#) builders, per CDNA5 ISA sec. 8.3 / 8.4.
// 2D tile from a row-major [tensor_d1 x tensor_d0] f16 tensor: groups 2/3 NULL.
// ---------------------------------------------------------------------------
__device__ __forceinline__ u32x4 tdm_group0(unsigned lds_addr,
                                            unsigned long long global_addr) {
  u32x4 g;
  g[0] = 1u;                                   // count=1, user descriptor
  g[1] = lds_addr;                             // LDS byte address (wave-rel.)
  g[2] = (unsigned)(global_addr & 0xffffffffu);
  g[3] = (unsigned)((global_addr >> 32) & 0x1ffffffu) | (2u << 30); // type=2
  return g;
}
__device__ __forceinline__ u32x8 tdm_group1(unsigned tensor_d0,
                                            unsigned tensor_d1,
                                            unsigned tile_d0,
                                            unsigned tile_d1,
                                            unsigned long long stride0) {
  u32x8 g;
  g[0] = 1u << 16;                                  // data_size = 2 bytes
  g[1] = (tensor_d0 & 0xffffu) << 16;               // tensor_dim0[15:0]
  g[2] = ((tensor_d0 >> 16) & 0xffffu) |            // tensor_dim0[31:16]
         ((tensor_d1 & 0xffffu) << 16);             // tensor_dim1[15:0]
  g[3] = ((tensor_d1 >> 16) & 0xffffu) |            // tensor_dim1[31:16]
         (tile_d0 << 16);                           // tile_dim0
  g[4] = tile_d1 & 0xffffu;                         // tile_dim1 (tile_dim2=0)
  g[5] = (unsigned)(stride0 & 0xffffffffu);         // tensor_dim0_stride lo
  g[6] = (unsigned)((stride0 >> 32) & 0xffffu);     // stride hi; dim1_stride=0
  g[7] = 0u;
  return g;
}

__device__ __forceinline__ void tdm_load_2d(unsigned lds_addr, const void* gptr,
                                            unsigned tensor_d0,
                                            unsigned tensor_d1,
                                            unsigned tile_d0, unsigned tile_d1,
                                            unsigned long long stride0) {
  u32x4 g0 = tdm_group0(lds_addr, (unsigned long long)(size_t)gptr);
  u32x8 g1 = tdm_group1(tensor_d0, tensor_d1, tile_d0, tile_d1, stride0);
  asm volatile("tensor_load_to_lds %0, %1" ::"s"(g0), "s"(g1) : "memory");
}

// ---------------------------------------------------------------------------
// Cross-lane reductions within 16-lane halves of a wave32 (masks keep bit4).
// ---------------------------------------------------------------------------
__device__ __forceinline__ float half16_max(float v) {
  v = fmaxf(v, __shfl_xor(v, 1));
  v = fmaxf(v, __shfl_xor(v, 2));
  v = fmaxf(v, __shfl_xor(v, 4));
  v = fmaxf(v, __shfl_xor(v, 8));
  return v;
}
__device__ __forceinline__ float half16_sum(float v) {
  v += __shfl_xor(v, 1);
  v += __shfl_xor(v, 2);
  v += __shfl_xor(v, 4);
  v += __shfl_xor(v, 8);
  return v;
}

// ---------------------------------------------------------------------------
// K0: fp32 -> f16 weight convert + transpose. W is [K x Ncol] row-major,
// output WT is [Ncol x K] so B-fragments read 16 contiguous halves per lane.
// ---------------------------------------------------------------------------
__global__ __launch_bounds__(256) void cvt_wT_kernel(
    const float* __restrict__ W, _Float16* __restrict__ WT, int K, int Ncol) {
  size_t i = (size_t)blockIdx.x * 256 + threadIdx.x;
  if (i >= (size_t)K * Ncol) return;
  int n = (int)(i / (size_t)K);
  int k = (int)(i - (size_t)n * K);
  WT[i] = (_Float16)W[(size_t)k * Ncol + n];
}

// ---------------------------------------------------------------------------
// K1: router. One block per batch (1024 threads = one token each).
// probs = softmax(x @ w_router). Expert-preferred routing: experts 2 then 1
// take top-256 (by prob, ties -> lower index) among untaken; rest -> expert 0.
// Rank-count selection reproduces sequential lax.top_k deterministically.
// ---------------------------------------------------------------------------
__global__ __launch_bounds__(1024) void router_kernel(
    const float* __restrict__ x, const float* __restrict__ wr,
    float* __restrict__ probs_out, int* __restrict__ mask_out,
    int* __restrict__ m_tok, float* __restrict__ p_sel) {
  __shared__ float sc2[1024];
  __shared__ float sc1[1024];
  __shared__ int   taken[1024];
  const int b = blockIdx.x;
  const int t = threadIdx.x;
  const float* row = x + ((size_t)b * 1024 + t) * 1024;
  float a0 = 0.f, a1 = 0.f, a2 = 0.f;
  for (int i = 0; i < 1024; ++i) {
    float xv = row[i];
    a0 += xv * wr[i * 3 + 0];
    a1 += xv * wr[i * 3 + 1];
    a2 += xv * wr[i * 3 + 2];
  }
  float mx = fmaxf(a0, fmaxf(a1, a2));
  float e0 = __expf(a0 - mx), e1 = __expf(a1 - mx), e2 = __expf(a2 - mx);
  float inv = 1.0f / (e0 + e1 + e2);
  float p0 = e0 * inv, p1 = e1 * inv, p2 = e2 * inv;
  size_t tg = (size_t)b * 1024 + t;
  probs_out[tg * 3 + 0] = p0;
  probs_out[tg * 3 + 1] = p1;
  probs_out[tg * 3 + 2] = p2;
  sc2[t] = p2; sc1[t] = p1;
  __syncthreads();
  int rank2 = 0;
  float my2 = sc2[t];
  for (int j = 0; j < 1024; ++j) {
    float o = sc2[j];
    rank2 += (o > my2) || (o == my2 && j < t);
  }
  int sel2 = rank2 < 256;
  taken[t] = sel2;
  __syncthreads();
  int rank1 = 0;
  float my1 = sc1[t];
  for (int j = 0; j < 1024; ++j) {
    float o = sc1[j];
    rank1 += (!taken[j]) && ((o > my1) || (o == my1 && j < t));
  }
  int sel1 = (!sel2) && (rank1 < 256);
  int mk = sel2 ? 2 : (sel1 ? 1 : 0);
  mask_out[tg] = mk;
  m_tok[tg]    = 256 << mk;                 // M_DIMS = (256,512,1024)
  p_sel[tg]    = (mk == 2) ? p2 : ((mk == 1) ? p1 : p0);
}

// ---------------------------------------------------------------------------
// K2: LayerNorm + nested channel mask -> f16. One block per token.
// ---------------------------------------------------------------------------
__global__ __launch_bounds__(256) void ln_mask_kernel(
    const float* __restrict__ in, const float* __restrict__ gamma,
    const float* __restrict__ beta, const int* __restrict__ m_tok,
    _Float16* __restrict__ out16) {
  __shared__ float r1[256];
  __shared__ float r2[256];
  const int tok = blockIdx.x;
  const int tid = threadIdx.x;
  const float* row = in + (size_t)tok * 1024;
  float v[4], s = 0.f, sq = 0.f;
  for (int i = 0; i < 4; ++i) {
    v[i] = row[tid + i * 256];
    s += v[i];
    sq += v[i] * v[i];
  }
  r1[tid] = s; r2[tid] = sq;
  __syncthreads();
  for (int off = 128; off; off >>= 1) {
    if (tid < off) { r1[tid] += r1[tid + off]; r2[tid] += r2[tid + off]; }
    __syncthreads();
  }
  const float mean = r1[0] * (1.0f / 1024.0f);
  const float var  = r2[0] * (1.0f / 1024.0f) - mean * mean;
  const float rstd = rsqrtf(var + 1e-5f);
  const int m = m_tok[tok];
  for (int i = 0; i < 4; ++i) {
    int idx = tid + i * 256;
    float val = (v[i] - mean) * rstd * gamma[idx] + beta[idx];
    out16[(size_t)tok * 1024 + idx] = (_Float16)((idx < m) ? val : 0.0f);
  }
}

// ---------------------------------------------------------------------------
// K3: block-tiled WMMA GEMM, double-buffered TDM pipeline.
// Block tile: 128 rows x 64 cols, 8 waves (wave w owns rows w*16..w*16+15,
// all 64 cols -> 4 f32 accumulators). K advances in 64-wide steps; wave 0
// issues the tensor_load_to_lds pair for step i+1 before the workgroup
// computes step i, waiting s_wait_tensorcnt<=2 (TENSORcnt completes in
// order, so step i's tiles have landed while step i+1's stay in flight).
// LDS: 2 x (A 128x64 + B 64x64) f16 = 48 KB of the 320 KB WGP pool.
// A row-major [M x Kd] f16; WT transposed weights [Nc x Kd] f16.
// Epilogues: 0: qkv (mask cols>=m, ->f16)
//            1: wo  (z = resid + (c+b)*mask, ->f32)
//            2: w1  (gelu(c+b)*hmask, mscale=4, ->f16)
//            3: w2  (out = resid + (alpha*p+1)*((c+b)*mask), ->f32)
// Fragments use k = 16*half + s in both A and B (contiguous, not the ISA table order).
// ---------------------------------------------------------------------------
__global__ __launch_bounds__(256) void gemm_wmma_tdm_kernel(
    const _Float16* __restrict__ A, const _Float16* __restrict__ WT,
    const float* __restrict__ bias, int M, int Nc, int Kd, int mode, int mscale,
    const int* __restrict__ m_tok, const float* __restrict__ resid,
    const float* __restrict__ p_sel, const float* __restrict__ alpha,
    _Float16* __restrict__ out16, float* __restrict__ out32) {
  __shared__ __align__(32) _Float16 sA[2][128 * 64];   // 2 x 16 KB
  __shared__ __align__(32) _Float16 sB[2][64 * 64];    // 2 x 8 KB
  const int lane = threadIdx.x & 31;
  const int wave = threadIdx.x >> 5;
  const int half = lane >> 4;
  const int l15  = lane & 15;
  const int tilesN = Nc >> 6;
  const int bm = blockIdx.x / tilesN;
  const int bn = blockIdx.x - bm * tilesN;

  // Generic->LDS: low 32 bits of a __shared__ pointer are the wave-relative
  // LDS byte address (flat-aperture truncation, ISA 10.2).
  const unsigned ldsA0 = (unsigned)(size_t)(void*)&sA[0][0];
  const unsigned ldsA1 = (unsigned)(size_t)(void*)&sA[1][0];
  const unsigned ldsB0 = (unsigned)(size_t)(void*)&sB[0][0];
  const unsigned ldsB1 = (unsigned)(size_t)(void*)&sB[1][0];

  const _Float16* Ab = A + (size_t)bm * 128 * Kd;
  const _Float16* Wb = WT + (size_t)bn * 64 * Kd;

  if (wave == 0) {  // prologue: stage K-step 0 into buffer 0
    tdm_load_2d(ldsA0, Ab, (unsigned)Kd, (unsigned)M, 64u, 128u,
                (unsigned long long)Kd);
    tdm_load_2d(ldsB0, Wb, (unsigned)Kd, (unsigned)Nc, 64u, 64u,
                (unsigned long long)Kd);
  }

  v8f acc[4] = {};
  int buf = 0;
  for (int k0 = 0; k0 < Kd; k0 += 64) {
    if (wave == 0) {
      const int kn = k0 + 64;
      if (kn < Kd) {  // prefetch next K-step into the other buffer
        tdm_load_2d(buf ? ldsA0 : ldsA1, Ab + kn, (unsigned)Kd, (unsigned)M,
                    64u, 128u, (unsigned long long)Kd);
        tdm_load_2d(buf ? ldsB0 : ldsB1, Wb + kn, (unsigned)Kd, (unsigned)Nc,
                    64u, 64u, (unsigned long long)Kd);
        __builtin_amdgcn_s_wait_tensorcnt(2);  // current step landed
      } else {
        __builtin_amdgcn_s_wait_tensorcnt(0);
      }
    }
    __syncthreads();  // publish current buffer
    const _Float16* aw = &sA[buf][(wave * 16 + l15) * 64 + half * 16];
    const _Float16* bw = &sB[buf][l15 * 64 + half * 16];
#pragma unroll
    for (int c = 0; c < 64; c += 32) {
      v8h alo = *(const v8h*)(aw + c);
      v8h ahi = *(const v8h*)(aw + c + 8);
      v16h af = SHUF16(alo, ahi);
#pragma unroll
      for (int nt = 0; nt < 4; ++nt) {
        v16h bf = *(const v16h*)(bw + nt * 16 * 64 + c);
        acc[nt] = __builtin_amdgcn_wmma_f32_16x16x32_f16(
            false, af, false, bf, (short)0, acc[nt], false, false);
      }
    }
    __syncthreads();  // all reads done before this buffer is re-filled
    buf ^= 1;
  }

  const float al = alpha ? alpha[0] : 0.0f;
  for (int nt = 0; nt < 4; ++nt) {
    const int col = bn * 64 + nt * 16 + l15;
    const float bv = bias ? bias[col] : 0.0f;
    for (int r = 0; r < 8; ++r) {
      const int row = bm * 128 + wave * 16 + half * 8 + r;  // token index
      const float keep = (col < m_tok[row] * mscale) ? 1.0f : 0.0f;
      const float c = acc[nt][r];
      if (mode == 0) {
        out16[(size_t)row * Nc + col] = (_Float16)(c * keep);
      } else if (mode == 1) {
        out32[(size_t)row * Nc + col] =
            resid[(size_t)row * Nc + col] + (c + bv) * keep;
      } else if (mode == 2) {
        float u = c + bv;
        float g = 0.5f * u * (1.0f + erff(u * 0.70710678f));  // exact GELU
        out16[(size_t)row * Nc + col] = (_Float16)(g * keep);
      } else {
        float zp = (c + bv) * keep;
        out32[(size_t)row * Nc + col] =
            resid[(size_t)row * Nc + col] + (al * p_sel[row] + 1.0f) * zp;
      }
    }
  }
}

// ---------------------------------------------------------------------------
// K4: transpose V per (b,h) head: vT[bh][d][token] so the P@V B-operand is
// key-contiguous. idx = bh<<16 | d<<10 | n.
// ---------------------------------------------------------------------------
__global__ __launch_bounds__(256) void transpose_v_kernel(
    const _Float16* __restrict__ vsrc, _Float16* __restrict__ vT) {
  int idx = blockIdx.x * 256 + threadIdx.x;
  int n  = idx & 1023;
  int d  = (idx >> 10) & 63;
  int bh = idx >> 16;
  int b = bh >> 4, h = bh & 15;
  vT[idx] = vsrc[((size_t)b * 1024 + n) * 1024 + h * 64 + d];
}

// ---------------------------------------------------------------------------
// K5: flash attention, one (b,h,64-query tile) per block, 4 waves, each wave
// owns 16 query rows. S = Q@K^T via WMMA; online softmax with half16 lane
// reductions; P re-laid out C-frag -> A-frag through LDS; O += P@V via WMMA.
// Masked heads (h*64 >= m_tok) write zeros (== o*dmask in the reference).
// ---------------------------------------------------------------------------
__global__ __launch_bounds__(128) void attn_flash_kernel(
    const _Float16* __restrict__ q16, const _Float16* __restrict__ k16,
    const _Float16* __restrict__ vT16, const int* __restrict__ m_tok,
    _Float16* __restrict__ o16) {
  __shared__ __align__(32) _Float16 plds[4][16 * 64];
  const int lane = threadIdx.x & 31;
  const int wave = threadIdx.x >> 5;
  const int half = lane >> 4;
  const int l15  = lane & 15;
  const int qt = blockIdx.x & 15;     // 16 query tiles of 64
  const int bh = blockIdx.x >> 4;     // b*16 + h
  const int b  = bh >> 4;
  const int h  = bh & 15;
  const int qrow0 = qt * 64 + wave * 16;
  const size_t tokbase = (size_t)b * 1024;

  const _Float16* qbase =
      q16 + (tokbase + qrow0 + l15) * 1024 + h * 64 + half * 16;

  float mrow[8], lrow[8];
  v8f oacc[4] = {};
  for (int r = 0; r < 8; ++r) { mrow[r] = -3.0e38f; lrow[r] = 0.0f; }

  for (int kt = 0; kt < 16; ++kt) {
    // ---- S tile: 16 query rows x 64 keys --------------------------------
    v8f s[4] = {};
    for (int c = 0; c < 64; c += 32) {
      v8h alo = *(const v8h*)(qbase + c);
      v8h ahi = *(const v8h*)(qbase + c + 8);
      v16h af = SHUF16(alo, ahi);
      for (int nt = 0; nt < 4; ++nt) {
        int key = kt * 64 + nt * 16 + l15;
        v16h bf = *(const v16h*)(k16 + (tokbase + key) * 1024 + h * 64 + c +
                                 half * 16);
        s[nt] = __builtin_amdgcn_wmma_f32_16x16x32_f16(
            false, af, false, bf, (short)0, s[nt], false, false);
      }
    }
    // ---- online softmax (scale = hd^-0.5 = 0.125) -----------------------
    float mnew[8], psum[8];
    for (int r = 0; r < 8; ++r) {
      float v = fmaxf(fmaxf(s[0][r], s[1][r]), fmaxf(s[2][r], s[3][r])) * 0.125f;
      mnew[r] = fmaxf(mrow[r], half16_max(v));
      psum[r] = 0.0f;
    }
    for (int nt = 0; nt < 4; ++nt) {
      for (int r = 0; r < 8; ++r) {
        float p = __expf(s[nt][r] * 0.125f - mnew[r]);
        psum[r] += p;
        plds[wave][(half * 8 + r) * 64 + nt * 16 + l15] = (_Float16)p;
      }
    }
    for (int r = 0; r < 8; ++r) {
      float sc = __expf(mrow[r] - mnew[r]);
      lrow[r] = lrow[r] * sc + half16_sum(psum[r]);
      mrow[r] = mnew[r];
      for (int nt = 0; nt < 4; ++nt) oacc[nt][r] *= sc;
    }
    __syncthreads();
    // ---- O += P @ V ------------------------------------------------------
    for (int c = 0; c < 64; c += 32) {
      const _Float16* pp = &plds[wave][(size_t)l15 * 64 + c + half * 16];
      v8h plo = *(const v8h*)pp;
      v8h phi = *(const v8h*)(pp + 8);
      v16h pf = SHUF16(plo, phi);
      for (int nt = 0; nt < 4; ++nt) {
        v16h bf = *(const v16h*)(vT16 + ((size_t)bh * 64 + nt * 16 + l15) * 1024 +
                                 kt * 64 + c + half * 16);
        oacc[nt] = __builtin_amdgcn_wmma_f32_16x16x32_f16(
            false, pf, false, bf, (short)0, oacc[nt], false, false);
      }
    }
    __syncthreads();
  }
  for (int r = 0; r < 8; ++r) {
    int tok = (int)tokbase + qrow0 + half * 8 + r;
    float invl = (h * 64 < m_tok[tok]) ? (1.0f / lrow[r]) : 0.0f;
    for (int nt = 0; nt < 4; ++nt) {
      o16[(size_t)tok * 1024 + h * 64 + nt * 16 + l15] =
          (_Float16)(oacc[nt][r] * invl);
    }
  }
}

// ---------------------------------------------------------------------------
// Host-side launcher.
// ---------------------------------------------------------------------------
extern "C" void kernel_launch(void* const* d_in, const int* in_sizes, int n_in,
                              void* d_out, int out_size, void* d_ws,
                              size_t ws_size, hipStream_t stream) {
  (void)in_sizes; (void)n_in; (void)out_size; (void)ws_size;
  const float* x     = (const float*)d_in[0];
  const float* wr    = (const float*)d_in[1];
  const float* ln1g  = (const float*)d_in[2];
  const float* ln1b  = (const float*)d_in[3];
  const float* wq    = (const float*)d_in[4];
  const float* wk    = (const float*)d_in[5];
  const float* wv    = (const float*)d_in[6];
  const float* wo    = (const float*)d_in[7];
  const float* bo    = (const float*)d_in[8];
  const float* ln2g  = (const float*)d_in[9];
  const float* ln2b  = (const float*)d_in[10];
  const float* w1    = (const float*)d_in[11];
  const float* b1    = (const float*)d_in[12];
  const float* w2    = (const float*)d_in[13];
  const float* b2    = (const float*)d_in[14];
  const float* alpha = (const float*)d_in[15];

  float* out_f   = (float*)d_out;                       // [B,N,D] fp32
  int*   mask_o  = (int*)(out_f + TOK * Dm);            // [B,N] int32
  float* probs_o = (float*)(mask_o + TOK);              // [B,N,3] fp32

  char* ws = (char*)d_ws;
  int*      m_tok = (int*)(ws + OFF_MTOK);
  float*    p_sel = (float*)(ws + OFF_PSEL);
  _Float16* h16   = (_Float16*)(ws + OFF_H16);
  _Float16* wqT   = (_Float16*)(ws + OFF_WQT);
  _Float16* wkT   = (_Float16*)(ws + OFF_WKT);
  _Float16* wvT   = (_Float16*)(ws + OFF_WVT);
  _Float16* woT   = (_Float16*)(ws + OFF_WOT);
  _Float16* w1T   = (_Float16*)(ws + OFF_W1T);
  _Float16* w2T   = (_Float16*)(ws + OFF_W2T);
  _Float16* q16   = (_Float16*)(ws + OFF_Q16);
  _Float16* k16   = (_Float16*)(ws + OFF_K16);
  _Float16* v16   = (_Float16*)(ws + OFF_V16);
  _Float16* vT16  = (_Float16*)(ws + OFF_VT16);
  _Float16* o16   = (_Float16*)(ws + OFF_O16);
  float*    zbuf  = (float*)(ws + OFF_Z);
  _Float16* h2_16 = (_Float16*)(ws + OFF_H2);
  _Float16* u16   = (_Float16*)(ws + OFF_U16);

  // 1) f16 transposed weights
  const int wg1 = (int)((Dm * Dm) / 256);
  cvt_wT_kernel<<<wg1, 256, 0, stream>>>(wq, wqT, 1024, 1024);
  cvt_wT_kernel<<<wg1, 256, 0, stream>>>(wk, wkT, 1024, 1024);
  cvt_wT_kernel<<<wg1, 256, 0, stream>>>(wv, wvT, 1024, 1024);
  cvt_wT_kernel<<<wg1, 256, 0, stream>>>(wo, woT, 1024, 1024);
  cvt_wT_kernel<<<wg1 * 4, 256, 0, stream>>>(w1, w1T, 1024, 4096);
  cvt_wT_kernel<<<wg1 * 4, 256, 0, stream>>>(w2, w2T, 4096, 1024);

  // 2) router: probs, expert mask, per-token nested dim, selected prob
  router_kernel<<<4, 1024, 0, stream>>>(x, wr, probs_o, mask_o, m_tok, p_sel);

  // 3) h = LN1(x) * dmask  (f16)
  ln_mask_kernel<<<TOK, 256, 0, stream>>>(x, ln1g, ln1b, m_tok, h16);

  // 4) Q,K,V = (h @ W) * dmask  (f16); 128x64 block tiles
  const int gemmD = (int)((TOK / 128) * (Dm / 64));        // 512 blocks
  gemm_wmma_tdm_kernel<<<gemmD, 256, 0, stream>>>(h16, wqT, nullptr, 4096,
      1024, 1024, 0, 1, m_tok, nullptr, nullptr, nullptr, q16, nullptr);
  gemm_wmma_tdm_kernel<<<gemmD, 256, 0, stream>>>(h16, wkT, nullptr, 4096,
      1024, 1024, 0, 1, m_tok, nullptr, nullptr, nullptr, k16, nullptr);
  gemm_wmma_tdm_kernel<<<gemmD, 256, 0, stream>>>(h16, wvT, nullptr, 4096,
      1024, 1024, 0, 1, m_tok, nullptr, nullptr, nullptr, v16, nullptr);

  // 5) per-head V transpose, then flash attention
  transpose_v_kernel<<<(int)(TOK * Dm / 256), 256, 0, stream>>>(v16, vT16);
  attn_flash_kernel<<<1024, 128, 0, stream>>>(q16, k16, vT16, m_tok, o16);

  // 6) z = x + ((o*dmask) @ wo + bo) * dmask   (fp32)
  gemm_wmma_tdm_kernel<<<gemmD, 256, 0, stream>>>(o16, woT, bo, 4096, 1024,
      1024, 1, 1, m_tok, x, nullptr, nullptr, nullptr, zbuf);

  // 7) h2 = LN2(z) * dmask  (f16)
  ln_mask_kernel<<<TOK, 256, 0, stream>>>(zbuf, ln2g, ln2b, m_tok, h2_16);

  // 8) u = gelu(h2 @ w1 + b1) * hmask  (f16)
  const int gemmF = (int)((TOK / 128) * (DFF / 64));       // 2048 blocks
  gemm_wmma_tdm_kernel<<<gemmF, 256, 0, stream>>>(h2_16, w1T, b1, 4096, 4096,
      1024, 2, 4, m_tok, nullptr, nullptr, nullptr, u16, nullptr);

  // 9) out = z + (alpha*p_sel + 1) * ((u @ w2 + b2) * dmask)  -> d_out
  gemm_wmma_tdm_kernel<<<gemmD, 256, 0, stream>>>(u16, w2T, b2, 4096, 1024,
      4096, 3, 1, m_tok, zbuf, p_sel, alpha, nullptr, out_f);
}
